// Sparse_Three_Concat_86973087744648
// MI455X (gfx1250) — matrix operand order repaired and checked by reading, not compiled, not run on hardware
//
#include <hip/hip_runtime.h>
#include <hip/hip_bf16.h>

// ---------------------------------------------------------------------------
// Types for CDNA5 WMMA (wave32): 16x16x32 bf16 -> f32
// ---------------------------------------------------------------------------
typedef __attribute__((ext_vector_type(16))) __bf16 bf16x16;
typedef __attribute__((ext_vector_type(8)))  __bf16 bf16x8;
typedef __attribute__((ext_vector_type(8)))  float  f32x8;

union BF16Frag { bf16x16 v; bf16x8 h[2]; };

static __device__ __forceinline__ unsigned short f2bf(float f) {
  union { float f; unsigned int u; } c; c.f = f;
  unsigned int u = c.u;
  unsigned int r = (u + 0x7FFFu + ((u >> 16) & 1u)) >> 16;  // round-nearest-even
  return (unsigned short)r;
}

// ---------------------------------------------------------------------------
// f32 -> bf16 raw conversion (contiguous)
// ---------------------------------------------------------------------------
__global__ void f32_to_bf16_kernel(const float* __restrict__ in,
                                   unsigned short* __restrict__ out, long n) {
  long i = (long)blockIdx.x * blockDim.x + threadIdx.x;
  if (i < n) out[i] = f2bf(in[i]);
}

// Transpose weight [K x Nw] f32 -> [Nw x K] bf16 (so WMMA B fragments are
// contiguous 32B loads per lane).
__global__ void prep_weight_kernel(const float* __restrict__ W,
                                   unsigned short* __restrict__ Wt, int K, int Nw) {
  long i = (long)blockIdx.x * blockDim.x + threadIdx.x;
  if (i >= (long)K * Nw) return;
  int k = (int)(i / Nw), n = (int)(i % Nw);
  Wt[(long)n * K + k] = f2bf(W[i]);
}

// out[r, colOff + j] = bias[j]  (pre-init scatter targets with bias)
__global__ void init_cols_bias_kernel(float* __restrict__ out, int ldOut, int colOff,
                                      const float* __restrict__ bias, int cols, int rows) {
  long i = (long)blockIdx.x * blockDim.x + threadIdx.x;
  if (i >= (long)rows * cols) return;
  int r = (int)(i / cols), j = (int)(i % cols);
  out[(long)r * ldOut + colOff + j] = bias[j];
}

// ---------------------------------------------------------------------------
// WMMA GEMM, 16x64 tile per wave: out = A[M x K](bf16) * W[K x Nw] + bias.
// Wt is pre-transposed [Nw x K]. One A fragment is reused across 4 B
// fragments / 4 f32 accumulators per k-step (4x less A traffic, 4 WMMAs per
// loop body). Nw must be a multiple of 64, K a multiple of 32, M of 16.
// ---------------------------------------------------------------------------
__global__ __launch_bounds__(256) void wmma_gemm_kernel(
    const unsigned short* __restrict__ A, const unsigned short* __restrict__ Wt,
    const float* __restrict__ bias, float* __restrict__ out,
    int M, int K, int Nw, int ldOut, int colOff)
{
  int wave   = blockIdx.x * (blockDim.x >> 5) + (threadIdx.x >> 5);
  int tilesN = Nw >> 6;                    // 64-wide N tiles
  int tiles  = (M >> 4) * tilesN;
  if (wave >= tiles) return;               // uniform per-wave: EXEC stays all-1s
  int tm = wave / tilesN, tn = wave % tilesN;
  int lane = threadIdx.x & 31;
  int half = lane >> 4, l16 = lane & 15;

  // A fragment, contiguous k order (not the ISA table order): lanes 0-15 row M=l16 hold K {0..15};
  // lanes 16-31 same rows hold K {16..31}.
  const unsigned short* arow = A  + (long)(tm * 16 + l16) * K + 16 * half;
  // B fragment, same contiguous k order as A: lane (l16, half) = column n=l16, contiguous K block.
  const unsigned short* brow = Wt + (long)(tn * 64 + l16) * K + 16 * half;
  const long bs = (long)16 * K;            // elements between 16-col subtiles

  f32x8 acc0 = {}, acc1 = {}, acc2 = {}, acc3 = {};
  for (int k0 = 0; k0 < K; k0 += 32) {
    __builtin_prefetch(arow + k0 + 128, 0, 1);   // speculative global_prefetch
    BF16Frag a;
    a.h[0] = *(const bf16x8*)(arow + k0);        // K = k0 + 16*half .. +7
    a.h[1] = *(const bf16x8*)(arow + k0 + 8);    // K = k0+8+16*half .. +7
    bf16x16 b0 = *(const bf16x16*)(brow + k0);
    bf16x16 b1 = *(const bf16x16*)(brow + bs + k0);
    bf16x16 b2 = *(const bf16x16*)(brow + 2 * bs + k0);
    bf16x16 b3 = *(const bf16x16*)(brow + 3 * bs + k0);
    acc0 = __builtin_amdgcn_wmma_f32_16x16x32_bf16(false, a.v, false, b0, (short)0, acc0, false, false);
    acc1 = __builtin_amdgcn_wmma_f32_16x16x32_bf16(false, a.v, false, b1, (short)0, acc1, false, false);
    acc2 = __builtin_amdgcn_wmma_f32_16x16x32_bf16(false, a.v, false, b2, (short)0, acc2, false, false);
    acc3 = __builtin_amdgcn_wmma_f32_16x16x32_bf16(false, a.v, false, b3, (short)0, acc3, false, false);
  }

  // C/D layout: VGPR r -> row (r + 8*half), col l16 (per 16-col subtile).
  float* orow = out + (long)(tm * 16 + 8 * half) * ldOut + colOff + tn * 64 + l16;
  f32x8 accs[4] = { acc0, acc1, acc2, acc3 };
#pragma unroll
  for (int j = 0; j < 4; ++j) {
    float bv = bias ? bias[tn * 64 + j * 16 + l16] : 0.0f;
#pragma unroll
    for (int r = 0; r < 8; ++r)
      orow[(long)r * ldOut + j * 16] = accs[j][r] + bv;
  }
}

// ---------------------------------------------------------------------------
// Edge scatter: out[dst, colOff + j] += ew * t[src, j]; 1 wave per edge,
// float4 gathers (b128) + 4 native f32 atomics per iteration.
// hcat target (150MB) is L2-resident -> atomics resolve in L2.
// ---------------------------------------------------------------------------
__global__ void scatter_kernel(const float* __restrict__ t, const int* __restrict__ src,
                               const int* __restrict__ dst, const float* __restrict__ ew,
                               float* __restrict__ out, int ldOut, int colOff, int cols, int E)
{
  long gid = (long)blockIdx.x * blockDim.x + threadIdx.x;
  int e = (int)(gid >> 5), lane = (int)(gid & 31);
  if (e >= E) return;
  int s = src[e], d = dst[e];
  float w = ew[e];
  const float4* tp4 = (const float4*)(t + (long)s * cols);
  float* op = out + (long)d * ldOut + colOff;
  int q = cols >> 2;                         // cols is a multiple of 4
  for (int j4 = lane; j4 < q; j4 += 32) {
    float4 v = tp4[j4];
    float* o = op + j4 * 4;
    unsafeAtomicAdd(o + 0, w * v.x);
    unsafeAtomicAdd(o + 1, w * v.y);
    unsafeAtomicAdd(o + 2, w * v.z);
    unsafeAtomicAdd(o + 3, w * v.w);
  }
}

// ---------------------------------------------------------------------------
// log_softmax over 64 columns; 1 wave per row, 2 cols per lane, shfl_xor tree.
// ---------------------------------------------------------------------------
__global__ void logsoftmax64_kernel(const float* __restrict__ in,
                                    float* __restrict__ out, int rows) {
  long gid = (long)blockIdx.x * blockDim.x + threadIdx.x;
  int row = (int)(gid >> 5), lane = (int)(gid & 31);
  if (row >= rows) return;
  const float* ip = in + (long)row * 64;
  float v0 = ip[lane], v1 = ip[lane + 32];
  float m = fmaxf(v0, v1);
  for (int off = 16; off; off >>= 1) m = fmaxf(m, __shfl_xor(m, off, 32));
  float e = expf(v0 - m) + expf(v1 - m);
  for (int off = 16; off; off >>= 1) e += __shfl_xor(e, off, 32);
  float ls = m + logf(e);
  float* op = out + (long)row * 64;
  op[lane] = v0 - ls;
  op[lane + 32] = v1 - ls;
}

// ---------------------------------------------------------------------------
// Host-side orchestration
// ---------------------------------------------------------------------------
static inline int g1(long n, int t) { return (int)((n + t - 1) / t); }

struct BlockW { const float *lnW,*lnb,*c1W,*c1b,*c2W,*c2b,*mixW,*mixb; };

static void run_block(unsigned short* xb, float* hcat, float* tmp, float* nxt,
                      unsigned short* wt, int Nn, int E, int di, int dout, int Hn,
                      const BlockW& w, const int* ei, const float* ew,
                      const int* ei2, const float* ew2, hipStream_t stream)
{
  const int T = 256;
  const int ld = 3 * dout;
  const int tiles  = (Nn / 16) * (dout / 64);   // 16x64 tiles per wave
  const int tiles2 = (Nn / 16) * (Hn / 64);

  // x0 = x @ lnW + lnb  -> hcat[:, 0:dout]
  prep_weight_kernel<<<g1((long)di * dout, T), T, 0, stream>>>(w.lnW, wt, di, dout);
  wmma_gemm_kernel<<<(tiles + 7) / 8, 256, 0, stream>>>(xb, wt, w.lnb, hcat, Nn, di, dout, ld, 0);

  // x1 = scatter(ei, ew, x @ c1W) + c1b -> hcat[:, dout:2*dout]
  prep_weight_kernel<<<g1((long)di * dout, T), T, 0, stream>>>(w.c1W, wt, di, dout);
  wmma_gemm_kernel<<<(tiles + 7) / 8, 256, 0, stream>>>(xb, wt, nullptr, tmp, Nn, di, dout, dout, 0);
  init_cols_bias_kernel<<<g1((long)Nn * dout, T), T, 0, stream>>>(hcat, ld, dout, w.c1b, dout, Nn);
  scatter_kernel<<<g1((long)E * 32, T), T, 0, stream>>>(tmp, ei, ei + E, ew, hcat, ld, dout, dout, E);

  // x2 = scatter(ei2, ew2, x @ c2W) + c2b -> hcat[:, 2*dout:3*dout]
  prep_weight_kernel<<<g1((long)di * dout, T), T, 0, stream>>>(w.c2W, wt, di, dout);
  wmma_gemm_kernel<<<(tiles + 7) / 8, 256, 0, stream>>>(xb, wt, nullptr, tmp, Nn, di, dout, dout, 0);
  init_cols_bias_kernel<<<g1((long)Nn * dout, T), T, 0, stream>>>(hcat, ld, 2 * dout, w.c2b, dout, Nn);
  scatter_kernel<<<g1((long)E * 32, T), T, 0, stream>>>(tmp, ei2, ei2 + E, ew2, hcat, ld, 2 * dout, dout, E);

  // mix: nxt = hcat @ mixW + mixb
  f32_to_bf16_kernel<<<g1((long)Nn * ld, T), T, 0, stream>>>(hcat, xb, (long)Nn * ld);
  prep_weight_kernel<<<g1((long)ld * Hn, T), T, 0, stream>>>(w.mixW, wt, ld, Hn);
  wmma_gemm_kernel<<<(tiles2 + 7) / 8, 256, 0, stream>>>(xb, wt, w.mixb, nxt, Nn, ld, Hn, Hn, 0);

  // next block input (bf16)
  f32_to_bf16_kernel<<<g1((long)Nn * Hn, T), T, 0, stream>>>(nxt, xb, (long)Nn * Hn);
}

extern "C" void kernel_launch(void* const* d_in, const int* in_sizes, int n_in,
                              void* d_out, int out_size, void* d_ws, size_t ws_size,
                              hipStream_t stream) {
  (void)n_in; (void)out_size; (void)ws_size;
  const float* x   = (const float*)d_in[0];
  const int*   ei  = (const int*)  d_in[1];
  const float* ew  = (const float*)d_in[2];
  const int*   ei2 = (const int*)  d_in[3];
  const float* ew2 = (const float*)d_in[4];

  const int Nn = in_sizes[0] / 256;   // 50000
  const int E  = in_sizes[1] / 2;     // 1600000

  // Workspace layout (256B aligned)
  size_t off = 0;
  auto alloc = [&](size_t bytes) { size_t o = off; off = (off + bytes + 255) & ~(size_t)255; return o; };
  char* ws = (char*)d_ws;
  unsigned short* xb   = (unsigned short*)(ws + alloc((size_t)Nn * 768 * sizeof(unsigned short)));
  float*          hcat = (float*)         (ws + alloc((size_t)Nn * 768 * sizeof(float)));
  float*          tmp  = (float*)         (ws + alloc((size_t)Nn * 256 * sizeof(float)));
  float*          nxt  = (float*)         (ws + alloc((size_t)Nn * 256 * sizeof(float)));
  unsigned short* wt   = (unsigned short*)(ws + alloc((size_t)768 * 256 * sizeof(unsigned short)));

  BlockW b1 = { (const float*)d_in[5],  (const float*)d_in[6],  (const float*)d_in[7],
                (const float*)d_in[8],  (const float*)d_in[9],  (const float*)d_in[10],
                (const float*)d_in[23], (const float*)d_in[24] };
  BlockW b2 = { (const float*)d_in[11], (const float*)d_in[12], (const float*)d_in[13],
                (const float*)d_in[14], (const float*)d_in[15], (const float*)d_in[16],
                (const float*)d_in[25], (const float*)d_in[26] };
  BlockW b3 = { (const float*)d_in[17], (const float*)d_in[18], (const float*)d_in[19],
                (const float*)d_in[20], (const float*)d_in[21], (const float*)d_in[22],
                (const float*)d_in[27], (const float*)d_in[28] };

  const int T = 256;
  f32_to_bf16_kernel<<<g1((long)Nn * 256, T), T, 0, stream>>>(x, xb, (long)Nn * 256);

  run_block(xb, hcat, tmp, nxt, wt, Nn, E, 256, 256, 256, b1, ei, ew, ei2, ew2, stream);
  run_block(xb, hcat, tmp, nxt, wt, Nn, E, 256, 256, 256, b2, ei, ew, ei2, ew2, stream);
  run_block(xb, hcat, tmp, nxt, wt, Nn, E, 256,  64,  64, b3, ei, ew, ei2, ew2, stream);

  // nxt holds [N, 64] logits
  logsoftmax64_kernel<<<g1((long)Nn * 32, T), T, 0, stream>>>(nxt, (float*)d_out, Nn);
}
